// Spline_Fitting_21328807592211
// MI455X (gfx1250) — matrix operand order repaired and checked by reading, not compiled, not run on hardware
//
#include <hip/hip_runtime.h>
#include <hip/hip_bf16.h>

// Problem constants (from reference): B=2, S=128, D=256, H=8, E=32, BH=16.
#define PB   2
#define PS   128
#define PD   256
#define PH   8
#define PE   32
#define PBH  (PB * PH)

typedef __attribute__((ext_vector_type(16))) _Float16 v16h;
typedef __attribute__((ext_vector_type(8)))  _Float16 v8h;
typedef __attribute__((ext_vector_type(8)))  float    v8f;

// ---------------------------------------------------------------------------
// WMMA fragment loaders (layouts per CDNA5 ISA 7.12.2, wave32).
//
// A (16x32 f16, row-major M x K in memory):
//   lane L: M = L&15, hw = L>>4.  Elements e=0..7  -> K = k0 + 8*hw + e
//                                 Elements e=8..15 -> K = k0 + 16 + 8*hw + (e-8)
//   => two contiguous 16-byte loads per lane.
//
// B (32x16 f16): lane L: N = L&15, hw = L>>4.  Same K mapping as A (8*hw + e, 16 + 8*hw + (e-8)).
//   With B supplied as K-contiguous per column (i.e. B^T row-major, which is
//   w_in/w_out natural [e,d] layout, or the xT buffer), this is two contiguous
//   16-byte loads per lane.
// ---------------------------------------------------------------------------
__device__ __forceinline__ v16h load_frag_a(const _Float16* A, int ldk, int row,
                                            int k0, int hw) {
  const _Float16* p = A + (size_t)row * ldk + k0 + 8 * hw;
  v8h lo = *(const v8h*)p;
  v8h hi = *(const v8h*)(p + 16);
  return __builtin_shufflevector(lo, hi, 0, 1, 2, 3, 4, 5, 6, 7,
                                 8, 9, 10, 11, 12, 13, 14, 15);
}

__device__ __forceinline__ v16h load_frag_b(const _Float16* colptr, int k0, int hw) {
  const _Float16* p = colptr + k0 + 8 * hw;
  v8h lo = *(const v8h*)p;
  v8h hi = *(const v8h*)(p + 16);
  return __builtin_shufflevector(lo, hi, 0, 1, 2, 3, 4, 5, 6, 7,
                                 8, 9, 10, 11, 12, 13, 14, 15);
}

// ---------------------------------------------------------------------------
// f32 -> f16 conversion (inputs, w_in, w_out)
// ---------------------------------------------------------------------------
__global__ void cvt_f16_kernel(const float* __restrict__ s,
                               _Float16* __restrict__ d, int n) {
  int i = blockIdx.x * blockDim.x + threadIdx.x;
  if (i < n) d[i] = (_Float16)s[i];
}

// ---------------------------------------------------------------------------
// GEMM1: x_pre[bh, s, e] = inputs[b,s,:] . w_in[e_out,:] + b_in[e_out]
// M = B*S = 256 rows, N = D = 256 cols, K = D = 256. One wave per 16x16 tile.
// Output is written directly into the (BH, S, E) permuted layout.
// ---------------------------------------------------------------------------
__global__ __launch_bounds__(32) void gemm_in_kernel(
    const _Float16* __restrict__ A, const _Float16* __restrict__ W,
    const float* __restrict__ bias, float* __restrict__ xout) {
  const int K = PD, NT = PD / 16;
  int tile = blockIdx.x;
  int tm = tile / NT, tn = tile % NT;
  int lane = threadIdx.x, hw = lane >> 4, l16 = lane & 15;
  int arow = tm * 16 + l16;
  int bcol = tn * 16 + l16;

  v8f acc = {};
  for (int k0 = 0; k0 < K; k0 += 32) {
    v16h a = load_frag_a(A, K, arow, k0, hw);
    v16h b = load_frag_b(W + (size_t)bcol * K, k0, hw);
    acc = __builtin_amdgcn_wmma_f32_16x16x32_f16(false, a, false, b,
                                                 (short)0, acc, false, false);
  }

  int eo = tn * 16 + l16;        // output feature 0..255
  int hh = eo >> 5;              // head
  int e  = eo & (PE - 1);        // within-head feature
  float bv = bias[eo];
#pragma unroll
  for (int r = 0; r < 8; ++r) {
    int m  = tm * 16 + 8 * hw + r;   // global row over B*S
    int bb = m >> 7;                  // / S
    int s  = m & (PS - 1);
    xout[(((size_t)bb * PH + hh) * PS + s) * PE + e] = acc[r] + bv;
  }
}

// ---------------------------------------------------------------------------
// Softmax over E (=32 = wave width) for each of BH*S rows, in place;
// also emit transposed f16 copy xT[bh, e, s] for the attn@x WMMA B-frags.
// ---------------------------------------------------------------------------
__global__ __launch_bounds__(256) void softmax_e_kernel(
    float* __restrict__ x, _Float16* __restrict__ xT) {
  int row  = (blockIdx.x * blockDim.x + threadIdx.x) >> 5;  // 0..BH*S-1
  int lane = threadIdx.x & 31;
  float v = x[(size_t)row * PE + lane];
  float m = v;
#pragma unroll
  for (int off = 16; off; off >>= 1) m = fmaxf(m, __shfl_xor(m, off, 32));
  float ev = __expf(v - m);
  float sum = ev;
#pragma unroll
  for (int off = 16; off; off >>= 1) sum += __shfl_xor(sum, off, 32);
  float y = ev / sum;
  x[(size_t)row * PE + lane] = y;
  int bh = row >> 7, s = row & (PS - 1);
  xT[((size_t)bh * PE + lane) * PS + s] = (_Float16)y;
}

// ---------------------------------------------------------------------------
// Spline scores + row softmax.
// attn_pre[i,j] = (sum_{p,q} [a*b*(1+0.5*min) - min^3/6] + E^2) / E
// Block = 256 threads (8 waves), handles 8 i-rows of one bh slice.
// x slice (128x32, pitch 33 to avoid LDS bank conflicts) staged in LDS.
// Each lane owns 4 j's; its j-rows are cached in 32 registers so the inner
// 1024-term loop is pure VALU (v_min_f32 / v_fma_f32). i-row values are
// broadcast across the wave with __shfl (wave-uniform row).
// ---------------------------------------------------------------------------
#define XPITCH 33
__global__ __launch_bounds__(256) void spline_attn_kernel(
    const float* __restrict__ x, _Float16* __restrict__ attn) {
  __shared__ float xs[PS * XPITCH];
  int bh = blockIdx.x >> 4;
  int rb = blockIdx.x & 15;
  const float* xb = x + (size_t)bh * PS * PE;
  for (int t = threadIdx.x; t < PS * PE; t += 256) {
    xs[(t >> 5) * XPITCH + (t & 31)] = xb[t];
  }
  __syncthreads();

  int wave = threadIdx.x >> 5;
  int lane = threadIdx.x & 31;
  int i = rb * 8 + wave;
  float xi = xs[i * XPITCH + lane];  // row i striped across lanes

  float sc[4];
#pragma unroll
  for (int jc = 0; jc < 4; ++jc) {
    int j = jc * 32 + lane;
    float brow[32];
#pragma unroll
    for (int q = 0; q < 32; ++q) brow[q] = xs[j * XPITCH + q];
    float acc = 0.0f;
    for (int p = 0; p < 32; ++p) {
      float a = __shfl(xi, p, 32);
#pragma unroll
      for (int q = 0; q < 32; ++q) {
        float b  = brow[q];
        float mn = fminf(a, b);
        float ab = a * b;
        acc = fmaf(ab, fmaf(0.5f, mn, 1.0f), acc);
        acc = fmaf(mn * mn * mn, -(1.0f / 6.0f), acc);
      }
    }
    sc[jc] = acc * (1.0f / (float)PE) + (float)PE;  // (+E^2)/E term
  }

  // softmax over j (128 values: 4 per lane x 32 lanes)
  float m = fmaxf(fmaxf(sc[0], sc[1]), fmaxf(sc[2], sc[3]));
#pragma unroll
  for (int off = 16; off; off >>= 1) m = fmaxf(m, __shfl_xor(m, off, 32));
  float e0 = __expf(sc[0] - m), e1 = __expf(sc[1] - m);
  float e2 = __expf(sc[2] - m), e3 = __expf(sc[3] - m);
  float sum = e0 + e1 + e2 + e3;
#pragma unroll
  for (int off = 16; off; off >>= 1) sum += __shfl_xor(sum, off, 32);
  float inv = 1.0f / sum;

  _Float16* arow = attn + ((size_t)bh * PS + i) * PS;
  arow[lane + 0]  = (_Float16)(e0 * inv);
  arow[lane + 32] = (_Float16)(e1 * inv);
  arow[lane + 64] = (_Float16)(e2 * inv);
  arow[lane + 96] = (_Float16)(e3 * inv);
}

// ---------------------------------------------------------------------------
// out_mid = attn @ x  (per bh: 128x128 . 128x32), via WMMA.
// B-frags come from xT (E x S per bh) so they are contiguous loads.
// Result written as f16 into (B*S, D) row-major (undoing the head permute).
// ---------------------------------------------------------------------------
__global__ __launch_bounds__(32) void gemm_attn_kernel(
    const _Float16* __restrict__ attn, const _Float16* __restrict__ xT,
    _Float16* __restrict__ mid) {
  int id = blockIdx.x;             // bh(16) * tm(8) * tn(2)
  int bh = id >> 4;
  int tm = (id >> 1) & 7;
  int tn = id & 1;
  const int K = PS;
  int lane = threadIdx.x, hw = lane >> 4, l16 = lane & 15;
  const _Float16* A = attn + (size_t)bh * PS * PS;
  const _Float16* X = xT + (size_t)bh * PE * PS;
  int arow = tm * 16 + l16;
  int bcol = tn * 16 + l16;        // e within head

  v8f acc = {};
  for (int k0 = 0; k0 < K; k0 += 32) {
    v16h a = load_frag_a(A, K, arow, k0, hw);
    v16h b = load_frag_b(X + (size_t)bcol * K, k0, hw);
    acc = __builtin_amdgcn_wmma_f32_16x16x32_f16(false, a, false, b,
                                                 (short)0, acc, false, false);
  }

  int b    = bh >> 3;
  int hh   = bh & 7;
  int dcol = hh * PE + tn * 16 + l16;
#pragma unroll
  for (int r = 0; r < 8; ++r) {
    int s = tm * 16 + 8 * hw + r;
    mid[((size_t)b * PS + s) * PD + dcol] = (_Float16)acc[r];
  }
}

// ---------------------------------------------------------------------------
// GEMM2: out[b,s,e] = mid[b,s,:] . w_out[e,:] + b_out[e]  (f32 output)
// ---------------------------------------------------------------------------
__global__ __launch_bounds__(32) void gemm_out_kernel(
    const _Float16* __restrict__ A, const _Float16* __restrict__ W,
    const float* __restrict__ bias, float* __restrict__ out) {
  const int K = PD, NT = PD / 16;
  int tile = blockIdx.x;
  int tm = tile / NT, tn = tile % NT;
  int lane = threadIdx.x, hw = lane >> 4, l16 = lane & 15;
  int arow = tm * 16 + l16;
  int bcol = tn * 16 + l16;

  v8f acc = {};
  for (int k0 = 0; k0 < K; k0 += 32) {
    v16h a = load_frag_a(A, K, arow, k0, hw);
    v16h b = load_frag_b(W + (size_t)bcol * K, k0, hw);
    acc = __builtin_amdgcn_wmma_f32_16x16x32_f16(false, a, false, b,
                                                 (short)0, acc, false, false);
  }

  int eo = tn * 16 + l16;
  float bv = bias[eo];
#pragma unroll
  for (int r = 0; r < 8; ++r) {
    int m = tm * 16 + 8 * hw + r;
    out[(size_t)m * PD + eo] = acc[r] + bv;
  }
}

// ---------------------------------------------------------------------------
// Launch
// ---------------------------------------------------------------------------
extern "C" void kernel_launch(void* const* d_in, const int* in_sizes, int n_in,
                              void* d_out, int out_size, void* d_ws, size_t ws_size,
                              hipStream_t stream) {
  (void)in_sizes; (void)n_in; (void)out_size; (void)ws_size;

  const float* inputs = (const float*)d_in[0];  // (B,S,D)   65536
  const float* w_in   = (const float*)d_in[1];  // (D,D)     65536
  const float* b_in   = (const float*)d_in[2];  // (D,)        256
  const float* w_out  = (const float*)d_in[3];  // (D,D)     65536
  const float* b_out  = (const float*)d_in[4];  // (D,)        256
  float* out = (float*)d_out;                   // (B,S,D)   65536

  // Workspace layout (all 256-byte aligned; total 1,441,792 bytes)
  char* ws = (char*)d_ws;
  _Float16* in_h   = (_Float16*)(ws + 0);        // 131072 B
  _Float16* win_h  = (_Float16*)(ws + 131072);   // 131072 B
  _Float16* wout_h = (_Float16*)(ws + 262144);   // 131072 B
  float*    x_f32  = (float*)   (ws + 393216);   // 262144 B (pre+post softmax)
  _Float16* xT_h   = (_Float16*)(ws + 655360);   // 131072 B (BH,E,S)
  _Float16* attn_h = (_Float16*)(ws + 786432);   // 524288 B (BH,S,S)
  _Float16* mid_h  = (_Float16*)(ws + 1310720);  // 131072 B (B*S,D)

  const int NTOT = PB * PS * PD;  // 65536

  cvt_f16_kernel<<<(NTOT + 255) / 256, 256, 0, stream>>>(inputs, in_h, NTOT);
  cvt_f16_kernel<<<(NTOT + 255) / 256, 256, 0, stream>>>(w_in, win_h, PD * PD);
  cvt_f16_kernel<<<(NTOT + 255) / 256, 256, 0, stream>>>(w_out, wout_h, PD * PD);

  gemm_in_kernel<<<(PB * PS / 16) * (PD / 16), 32, 0, stream>>>(in_h, win_h, b_in, x_f32);

  softmax_e_kernel<<<(PBH * PS) / 8, 256, 0, stream>>>(x_f32, xT_h);

  spline_attn_kernel<<<PBH * (PS / 8), 256, 0, stream>>>(x_f32, attn_h);

  gemm_attn_kernel<<<PBH * (PS / 16) * (PE / 16), 32, 0, stream>>>(attn_h, xT_h, mid_h);

  gemm_out_kernel<<<(PB * PS / 16) * (PD / 16), 32, 0, stream>>>(mid_h, wout_h, b_out, out);
}
